// AttentionHead_9723805958414
// MI455X (gfx1250) — matrix operand order repaired and checked by reading, not compiled, not run on hardware
//
#include <hip/hip_runtime.h>
#include <hip/hip_bf16.h>

#define NPTS 100000
#define KNBR 16
#define DIM  64
#define BN_EPS 1e-5f

typedef __attribute__((ext_vector_type(16))) _Float16 v16h;
typedef __attribute__((ext_vector_type(8)))  float    v8f;

// A-matrix (16-bit, 16x32): lane l holds row l&15; element j -> K channel:
__device__ __forceinline__ int chanA(int j, int kh) { return j + 8 * kh + (j >= 8 ? 8 : 0); }
// B-matrix (16-bit, 32x16): lane l holds col l&15; element j -> K channel:
__device__ __forceinline__ int chanB(int j, int kh) { return j + 8 * kh + (j >= 8 ? 8 : 0); }

#define WMMA_F16(A, B, C) \
  __builtin_amdgcn_wmma_f32_16x16x32_f16(false, (A), false, (B), (short)0, (C), false, false)

// ---------------------------------------------------------------------------
// Kernel 1: q/k/v GEMMs  [N,64] x [64,64]^T + bias  via f16 WMMA, f32 accum.
// One wave handles a 16-row tile across all 64 cols for all 3 weight sets.
// ---------------------------------------------------------------------------
__global__ __launch_bounds__(256) void qkv_kernel(
    const float* __restrict__ feats,
    const float* __restrict__ qw, const float* __restrict__ qb,
    const float* __restrict__ kw, const float* __restrict__ kb,
    const float* __restrict__ vw, const float* __restrict__ vb,
    float* __restrict__ qo, float* __restrict__ ko, float* __restrict__ vo)
{
  int wid = (blockIdx.x * blockDim.x + threadIdx.x) >> 5;
  if (wid >= NPTS / 16) return;
  int lane = threadIdx.x & 31;
  int r = lane & 15, kh = lane >> 4;
  int row = wid * 16 + r;
  const float* frow = feats + row * DIM;

  v16h a_lo, a_hi;
#pragma unroll
  for (int j = 0; j < 16; ++j) {
    int c = chanA(j, kh);
    a_lo[j] = (_Float16)frow[c];
    a_hi[j] = (_Float16)frow[c + 32];
  }

  const float* Ws[3] = {qw, kw, vw};
  const float* Bs[3] = {qb, kb, vb};
  float*       Os[3] = {qo, ko, vo};

#pragma unroll
  for (int m = 0; m < 3; ++m) {
#pragma unroll
    for (int t = 0; t < 4; ++t) {
      int col = t * 16 + r;
      const float* wrow = Ws[m] + col * DIM;
      v16h b_lo, b_hi;
#pragma unroll
      for (int j = 0; j < 16; ++j) {
        int c = chanB(j, kh);
        b_lo[j] = (_Float16)wrow[c];
        b_hi[j] = (_Float16)wrow[c + 32];
      }
      float bias = Bs[m][col];
      v8f acc;
#pragma unroll
      for (int i = 0; i < 8; ++i) acc[i] = bias;
      acc = WMMA_F16(a_lo, b_lo, acc);
      acc = WMMA_F16(a_hi, b_hi, acc);
      float* out = Os[m];
#pragma unroll
      for (int i = 0; i < 8; ++i)
        out[(wid * 16 + kh * 8 + i) * DIM + col] = acc[i];   // C layout: row = i + 8*kh
    }
  }
}

// ---------------------------------------------------------------------------
// Kernel 2: per-channel sum / sum-sq of p_lin = p_g @ p_w.T + p_b (K=3 MLP).
// Wave iterates (n,k) pairs; each lane owns channels {l, l+32}.
// ---------------------------------------------------------------------------
__global__ __launch_bounds__(256) void pstats_kernel(
    const int* __restrict__ nbr, const float* __restrict__ pts,
    const float* __restrict__ pw, const float* __restrict__ pb,
    float* __restrict__ gsum, float* __restrict__ gsq)
{
  __shared__ float s_sum[DIM], s_sq[DIM];
  if (threadIdx.x < DIM) { s_sum[threadIdx.x] = 0.f; s_sq[threadIdx.x] = 0.f; }
  __syncthreads();

  int lane = threadIdx.x & 31;
  int c0 = lane, c1 = lane + 32;
  float w00 = pw[c0 * 3 + 0], w01 = pw[c0 * 3 + 1], w02 = pw[c0 * 3 + 2], bb0 = pb[c0];
  float w10 = pw[c1 * 3 + 0], w11 = pw[c1 * 3 + 1], w12 = pw[c1 * 3 + 2], bb1 = pb[c1];

  int gw = (blockIdx.x * blockDim.x + threadIdx.x) >> 5;
  int nw = (gridDim.x * blockDim.x) >> 5;
  float sum0 = 0.f, sq0 = 0.f, sum1 = 0.f, sq1 = 0.f;

  for (int p = gw; p < NPTS * KNBR; p += nw) {
    int idx = nbr[p];
    float px, py, pz;
    if (idx >= NPTS) { px = py = pz = 1.0e6f; }
    else { px = pts[idx * 3 + 0]; py = pts[idx * 3 + 1]; pz = pts[idx * 3 + 2]; }
    float v0 = fmaf(px, w00, fmaf(py, w01, fmaf(pz, w02, bb0)));
    float v1 = fmaf(px, w10, fmaf(py, w11, fmaf(pz, w12, bb1)));
    sum0 += v0; sq0 += v0 * v0;
    sum1 += v1; sq1 += v1 * v1;
  }
  atomicAdd(&s_sum[c0], sum0); atomicAdd(&s_sq[c0], sq0);
  atomicAdd(&s_sum[c1], sum1); atomicAdd(&s_sq[c1], sq1);
  __syncthreads();
  if (threadIdx.x < DIM) {
    atomicAdd(&gsum[threadIdx.x], s_sum[threadIdx.x]);
    atomicAdd(&gsq[threadIdx.x],  s_sq[threadIdx.x]);
  }
}

// ---------------------------------------------------------------------------
// BN finalize: a = gamma*rsqrt(var+eps), b = beta - mean*a   (64 threads)
// ---------------------------------------------------------------------------
__global__ void bnfin_kernel(const float* __restrict__ sum, const float* __restrict__ sq,
                             const float* __restrict__ gamma, const float* __restrict__ beta,
                             float* __restrict__ a, float* __restrict__ b)
{
  int c = threadIdx.x;
  const float invn = 1.0f / (float)(NPTS * KNBR);
  float mean = sum[c] * invn;
  float var  = sq[c] * invn - mean * mean;
  float s    = gamma[c] * rsqrtf(var + BN_EPS);
  a[c] = s;
  b[c] = beta[c] - mean * s;
}

// ---------------------------------------------------------------------------
// Shared helpers for the w-GEMM passes (one wave per point; rows = neighbors)
// ---------------------------------------------------------------------------
__device__ __forceinline__ void load_b_frags(const float* __restrict__ ww, int lane,
                                             v16h* b_lo, v16h* b_hi)
{
  int r = lane & 15, kh = lane >> 4;
#pragma unroll
  for (int t = 0; t < 4; ++t) {
    const float* wrow = ww + (t * 16 + r) * DIM;
#pragma unroll
    for (int j = 0; j < 16; ++j) {
      int c = chanB(j, kh);
      b_lo[t][j] = (_Float16)wrow[c];
      b_hi[t][j] = (_Float16)wrow[c + 32];
    }
  }
}

// Build A fragment: w_pre[r,c] = k_g[r,c]*q[n,c] + relu(BN(p_lin[r,c]))
__device__ __forceinline__ void build_a_frag(
    int n, int lane,
    const int* __restrict__ nbr, const float* __restrict__ pts,
    const float* __restrict__ kbuf, const float* __restrict__ qbuf,
    const float* s_pw, const float* s_pb, const float* s_ap, const float* s_bp,
    v16h& a_lo, v16h& a_hi)
{
  int r = lane & 15, kh = lane >> 4;
  int idx = nbr[n * KNBR + r];
  bool pad = (idx >= NPTS);
  float px, py, pz;
  if (pad) { px = py = pz = 1.0e6f; }
  else { px = pts[idx * 3 + 0]; py = pts[idx * 3 + 1]; pz = pts[idx * 3 + 2]; }
  const float* krow = kbuf + (pad ? 0 : idx) * DIM;
  const float* qrow = qbuf + n * DIM;
#pragma unroll
  for (int j = 0; j < 16; ++j) {
    int c = chanA(j, kh);
    {
      float pl = fmaf(px, s_pw[c * 3], fmaf(py, s_pw[c * 3 + 1], fmaf(pz, s_pw[c * 3 + 2], s_pb[c])));
      float pf = fmaxf(0.f, fmaf(s_ap[c], pl, s_bp[c]));
      float kg = pad ? 0.f : krow[c];
      a_lo[j] = (_Float16)fmaf(kg, qrow[c], pf);
    }
    int ch = c + 32;
    {
      float pl = fmaf(px, s_pw[ch * 3], fmaf(py, s_pw[ch * 3 + 1], fmaf(pz, s_pw[ch * 3 + 2], s_pb[ch])));
      float pf = fmaxf(0.f, fmaf(s_ap[ch], pl, s_bp[ch]));
      float kg = pad ? 0.f : krow[ch];
      a_hi[j] = (_Float16)fmaf(kg, qrow[ch], pf);
    }
  }
}

// ---------------------------------------------------------------------------
// Kernel 3: per-channel stats of w_lin (pre-BN) via WMMA, recomputed.
// ---------------------------------------------------------------------------
__global__ __launch_bounds__(256) void wstats_kernel(
    const int* __restrict__ nbr, const float* __restrict__ pts,
    const float* __restrict__ kbuf, const float* __restrict__ qbuf,
    const float* __restrict__ p_w, const float* __restrict__ p_b,
    const float* __restrict__ a_p, const float* __restrict__ b_p,
    const float* __restrict__ w_w, const float* __restrict__ w_b,
    float* __restrict__ gsum, float* __restrict__ gsq)
{
  __shared__ float s_pw[DIM * 3], s_pb[DIM], s_ap[DIM], s_bp[DIM], s_wb[DIM];
  __shared__ float s_sum[DIM], s_sq[DIM];
  for (int i = threadIdx.x; i < DIM * 3; i += blockDim.x) s_pw[i] = p_w[i];
  if (threadIdx.x < DIM) {
    s_pb[threadIdx.x] = p_b[threadIdx.x];
    s_ap[threadIdx.x] = a_p[threadIdx.x];
    s_bp[threadIdx.x] = b_p[threadIdx.x];
    s_wb[threadIdx.x] = w_b[threadIdx.x];
    s_sum[threadIdx.x] = 0.f; s_sq[threadIdx.x] = 0.f;
  }
  __syncthreads();

  int lane = threadIdx.x & 31;
  int r = lane & 15;
  v16h b_lo[4], b_hi[4];
  load_b_frags(w_w, lane, b_lo, b_hi);     // weights resident in VGPRs

  float tsum[4] = {0.f, 0.f, 0.f, 0.f}, tsq[4] = {0.f, 0.f, 0.f, 0.f};
  int gw = (blockIdx.x * blockDim.x + threadIdx.x) >> 5;
  int nw = (gridDim.x * blockDim.x) >> 5;

  for (int n = gw; n < NPTS; n += nw) {
    v16h a_lo, a_hi;
    build_a_frag(n, lane, nbr, pts, kbuf, qbuf, s_pw, s_pb, s_ap, s_bp, a_lo, a_hi);
#pragma unroll
    for (int t = 0; t < 4; ++t) {
      float bias = s_wb[t * 16 + r];
      v8f acc;
#pragma unroll
      for (int i = 0; i < 8; ++i) acc[i] = bias;
      acc = WMMA_F16(a_lo, b_lo[t], acc);
      acc = WMMA_F16(a_hi, b_hi[t], acc);
#pragma unroll
      for (int i = 0; i < 8; ++i) { float v = acc[i]; tsum[t] += v; tsq[t] += v * v; }
    }
  }
#pragma unroll
  for (int t = 0; t < 4; ++t) {
    tsum[t] += __shfl_xor(tsum[t], 16, 32);   // lane l and l^16 share the same column
    tsq[t]  += __shfl_xor(tsq[t],  16, 32);
  }
  if (lane < 16) {
#pragma unroll
    for (int t = 0; t < 4; ++t) {
      atomicAdd(&s_sum[t * 16 + lane], tsum[t]);
      atomicAdd(&s_sq [t * 16 + lane], tsq[t]);
    }
  }
  __syncthreads();
  if (threadIdx.x < DIM) {
    atomicAdd(&gsum[threadIdx.x], s_sum[threadIdx.x]);
    atomicAdd(&gsq[threadIdx.x],  s_sq[threadIdx.x]);
  }
}

// ---------------------------------------------------------------------------
// Kernel 4: final pass — WMMA logits, BN+ReLU, softmax over the 16 neighbor
// rows (in-register + one shfl_xor(16)), then weighted sum of (v_g + p_feats).
// ---------------------------------------------------------------------------
__global__ __launch_bounds__(256) void final_kernel(
    const int* __restrict__ nbr, const float* __restrict__ pts,
    const float* __restrict__ kbuf, const float* __restrict__ qbuf,
    const float* __restrict__ vbuf,
    const float* __restrict__ p_w, const float* __restrict__ p_b,
    const float* __restrict__ a_p, const float* __restrict__ b_p,
    const float* __restrict__ w_w, const float* __restrict__ w_b,
    const float* __restrict__ a_w, const float* __restrict__ b_w,
    float* __restrict__ out)
{
  __shared__ float s_pw[DIM * 3], s_pb[DIM], s_ap[DIM], s_bp[DIM],
                   s_wb[DIM], s_aw[DIM], s_bw[DIM];
  for (int i = threadIdx.x; i < DIM * 3; i += blockDim.x) s_pw[i] = p_w[i];
  if (threadIdx.x < DIM) {
    s_pb[threadIdx.x] = p_b[threadIdx.x];
    s_ap[threadIdx.x] = a_p[threadIdx.x];
    s_bp[threadIdx.x] = b_p[threadIdx.x];
    s_wb[threadIdx.x] = w_b[threadIdx.x];
    s_aw[threadIdx.x] = a_w[threadIdx.x];
    s_bw[threadIdx.x] = b_w[threadIdx.x];
  }
  __syncthreads();

  int lane = threadIdx.x & 31;
  int r = lane & 15, kh = lane >> 4;
  v16h b_lo[4], b_hi[4];
  load_b_frags(w_w, lane, b_lo, b_hi);

  int gw = (blockIdx.x * blockDim.x + threadIdx.x) >> 5;
  int nw = (gridDim.x * blockDim.x) >> 5;

  for (int n = gw; n < NPTS; n += nw) {
    v16h a_lo, a_hi;
    build_a_frag(n, lane, nbr, pts, kbuf, qbuf, s_pw, s_pb, s_ap, s_bp, a_lo, a_hi);

    // Neighbor data for this lane's 8 C-layout rows (row = kh*8 + i)
    int   ridx[8];
    float rpx[8], rpy[8], rpz[8];
#pragma unroll
    for (int i = 0; i < 8; ++i) {
      int idx = nbr[n * KNBR + kh * 8 + i];
      ridx[i] = idx;
      if (idx >= NPTS) { rpx[i] = rpy[i] = rpz[i] = 1.0e6f; }
      else { rpx[i] = pts[idx * 3 + 0]; rpy[i] = pts[idx * 3 + 1]; rpz[i] = pts[idx * 3 + 2]; }
    }

#pragma unroll
    for (int t = 0; t < 4; ++t) {
      int col = t * 16 + r;
      float bias = s_wb[col];
      v8f acc;
#pragma unroll
      for (int i = 0; i < 8; ++i) acc[i] = bias;
      acc = WMMA_F16(a_lo, b_lo[t], acc);
      acc = WMMA_F16(a_hi, b_hi[t], acc);

      float aw = s_aw[col], bw = s_bw[col];
      float y[8];
      float m = -1e30f;
#pragma unroll
      for (int i = 0; i < 8; ++i) { y[i] = fmaxf(0.f, fmaf(aw, acc[i], bw)); m = fmaxf(m, y[i]); }
      m = fmaxf(m, __shfl_xor(m, 16, 32));           // max over all 16 neighbor rows
      float s = 0.f;
#pragma unroll
      for (int i = 0; i < 8; ++i) { y[i] = __expf(y[i] - m); s += y[i]; }
      s += __shfl_xor(s, 16, 32);
      float inv = 1.0f / s;

      float pw0 = s_pw[col * 3], pw1 = s_pw[col * 3 + 1], pw2 = s_pw[col * 3 + 2];
      float pbc = s_pb[col], apc = s_ap[col], bpc = s_bp[col];
      float accum = 0.f;
#pragma unroll
      for (int i = 0; i < 8; ++i) {
        int idx = ridx[i];
        bool pad = idx >= NPTS;
        float pl = fmaf(rpx[i], pw0, fmaf(rpy[i], pw1, fmaf(rpz[i], pw2, pbc)));
        float pf = fmaxf(0.f, fmaf(apc, pl, bpc));
        float vg = pad ? 0.f : vbuf[idx * DIM + col];
        accum += (vg + pf) * (y[i] * inv);
      }
      accum += __shfl_xor(accum, 16, 32);            // combine the two row halves
      if (lane < 16) out[n * DIM + col] = accum;
    }
  }
}

// ---------------------------------------------------------------------------
extern "C" void kernel_launch(void* const* d_in, const int* in_sizes, int n_in,
                              void* d_out, int out_size, void* d_ws, size_t ws_size,
                              hipStream_t stream)
{
  (void)in_sizes; (void)n_in; (void)out_size; (void)ws_size;
  const float* points    = (const float*)d_in[0];
  const int*   neighbors = (const int*)  d_in[1];
  const float* feats     = (const float*)d_in[2];
  const float* q_w = (const float*)d_in[3];  const float* q_b = (const float*)d_in[4];
  const float* k_w = (const float*)d_in[5];  const float* k_b = (const float*)d_in[6];
  const float* v_w = (const float*)d_in[7];  const float* v_b = (const float*)d_in[8];
  const float* p_w = (const float*)d_in[9];  const float* p_b = (const float*)d_in[10];
  const float* p_gamma = (const float*)d_in[11]; const float* p_beta = (const float*)d_in[12];
  const float* w_w = (const float*)d_in[13]; const float* w_b = (const float*)d_in[14];
  const float* w_gamma = (const float*)d_in[15]; const float* w_beta = (const float*)d_in[16];
  float* out = (float*)d_out;

  float* ws    = (float*)d_ws;
  float* qbuf  = ws;                       // N*64
  float* kbuf  = qbuf + (size_t)NPTS * DIM;
  float* vbuf  = kbuf + (size_t)NPTS * DIM;
  float* stats = vbuf + (size_t)NPTS * DIM; // 512 floats
  float* p_sum = stats + 0;
  float* p_sq  = stats + 64;
  float* a_p   = stats + 128;
  float* b_p   = stats + 192;
  float* w_sum = stats + 256;
  float* w_sq  = stats + 320;
  float* a_w   = stats + 384;
  float* b_w   = stats + 448;

  hipMemsetAsync(stats, 0, 512 * sizeof(float), stream);

  {
    int waves  = NPTS / 16;               // 6250 row-tiles
    int blocks = (waves + 7) / 8;         // 8 waves / 256-thread block
    qkv_kernel<<<blocks, 256, 0, stream>>>(feats, q_w, q_b, k_w, k_b, v_w, v_b,
                                           qbuf, kbuf, vbuf);
  }
  pstats_kernel<<<1024, 256, 0, stream>>>(neighbors, points, p_w, p_b, p_sum, p_sq);
  bnfin_kernel<<<1, 64, 0, stream>>>(p_sum, p_sq, p_gamma, p_beta, a_p, b_p);
  wstats_kernel<<<512, 256, 0, stream>>>(neighbors, points, kbuf, qbuf,
                                         p_w, p_b, a_p, b_p, w_w, w_b, w_sum, w_sq);
  bnfin_kernel<<<1, 64, 0, stream>>>(w_sum, w_sq, w_gamma, w_beta, a_w, b_w);
  final_kernel<<<512, 256, 0, stream>>>(neighbors, points, kbuf, qbuf, vbuf,
                                        p_w, p_b, a_p, b_p, w_w, w_b, a_w, b_w, out);
}
